// SparseMultiHeadAttention_51685636440099
// MI455X (gfx1250) — compile-verified, matrix operand order checked by reading, not run on hardware
//
#include <hip/hip_runtime.h>
#include <hip/hip_bf16.h>

// ---------------------------------------------------------------------------
// SparseMultiHeadAttention for MI455X (gfx1250), wave32 + WMMA bf16 path.
//   B=8, S=4096, H=512, num_heads=64, t=64.
//   ~73 GFLOP, ~330 MB traffic with bf16 intermediates -> memory-roofline
//   bound (~14 us at 23.3 TB/s); bf16 WMMA (16x16x32) keeps compute off the
//   critical path vs. the 8x-slower fp32 WMMA.
// ---------------------------------------------------------------------------

typedef __attribute__((ext_vector_type(16))) __bf16 v16bf;
typedef __attribute__((ext_vector_type(8)))  float  v8f;

namespace {

constexpr int BB = 8;      // batch
constexpr int SS = 4096;   // sequence
constexpr int HH = 512;    // hidden
constexpr int NH = 64;     // heads
constexpr int TT = 64;     // tokens per head

constexpr size_t WELT = (size_t)HH * HH;            // 262144 elems per weight
constexpr size_t QELT = (size_t)BB * SS * HH;       // 16777216 elems

__device__ __forceinline__ unsigned short f2bf(float f) {
  union { float f; unsigned u; } x; x.f = f;
  unsigned u = x.u;
  unsigned r = u + 0x7fffu + ((u >> 16) & 1u);   // round-to-nearest-even
  return (unsigned short)(r >> 16);
}

union FragBF { v16bf v; uint4 q[2]; unsigned short s[16]; };

// A/B fragment from row-major bf16, row stride ld (elems).
// A: base -> (m0,k0), lane&15 = row. B: base -> (n0*ld + k0), lane&15 = col
// (same load because our B columns are rows of W / rows of K in memory).
// Layout per ISA: lanes 0-15 hold K 0..7 (VGPR0-3) and 16..23 (VGPR4-7);
// lanes 16-31 hold K 8..15 and 24..31.
__device__ __forceinline__ v16bf load_frag16(const unsigned short* base,
                                             int ld, int lane) {
  FragBF f;
  const int half = (lane >> 4) & 1;
  const int r    = lane & 15;
  const unsigned short* p = base + (size_t)r * ld + half * 8;
  f.q[0] = *(const uint4*)(p);
  f.q[1] = *(const uint4*)(p + 16);
  return f.v;
}

// B fragment where the matrix is stored row-major but we need true columns
// (used for P @ V : B[k][n] = V[k][n], column n is strided by ld).
__device__ __forceinline__ v16bf load_frag16_strided(const unsigned short* base,
                                                     int ld, int lane) {
  FragBF f;
  const int half = (lane >> 4) & 1;
  const int r    = lane & 15;
  const unsigned short* p = base + r;
#pragma unroll
  for (int j = 0; j < 8; ++j) {
    f.s[j]     = p[(size_t)(half * 8 + j) * ld];
    f.s[8 + j] = p[(size_t)(16 + half * 8 + j) * ld];
  }
  return f.v;
}

__device__ __forceinline__ v8f wmma_bf16(v16bf a, v16bf b, v8f c) {
  return __builtin_amdgcn_wmma_f32_16x16x32_bf16(false, a, false, b,
                                                 (short)0, c, false, false);
}

// ---------------------------------------------------------------------------
// Kernel 0: fp32 -> bf16 weight conversion (Wq,Wk,Wv,Wo).
// ---------------------------------------------------------------------------
__global__ __launch_bounds__(256) void k_convert_weights(
    const float* __restrict__ wq, const float* __restrict__ wk,
    const float* __restrict__ wv, const float* __restrict__ wo,
    unsigned short* __restrict__ dst) {
  const float* srcs[4] = {wq, wk, wv, wo};
  const float* s = srcs[blockIdx.y];
  unsigned short* d = dst + (size_t)blockIdx.y * WELT;
  for (int i = blockIdx.x * blockDim.x + threadIdx.x; i < (int)WELT;
       i += gridDim.x * blockDim.x)
    d[i] = f2bf(s[i]);
}

// ---------------------------------------------------------------------------
// Kernel 1: fused QKV projection with permutation gather.
//   Qp[b, i, :] = x[b, perm[i], :] @ Wq^T + bq   (bf16 out), same for K, V.
//   grid (S/32, H/64, B), 256 threads = 8 waves (2 M-tiles x 4 N-tiles).
// ---------------------------------------------------------------------------
constexpr int LDA = HH + 8;   // 520 elems; 1040 B row stride (16B aligned)

__global__ __launch_bounds__(256) void k_qkv_project(
    const float* __restrict__ x,
    const unsigned short* __restrict__ wq_bf,
    const unsigned short* __restrict__ wk_bf,
    const unsigned short* __restrict__ wv_bf,
    const float* __restrict__ bq, const float* __restrict__ bk,
    const float* __restrict__ bv, const int* __restrict__ perm,
    unsigned short* __restrict__ qp, unsigned short* __restrict__ kp,
    unsigned short* __restrict__ vp) {
  __shared__ unsigned short xs[32 * LDA];   // 32 gathered rows of x as bf16

  const int tid  = threadIdx.x;
  const int wave = tid >> 5;
  const int lane = tid & 31;
  const int wm   = wave & 1;        // 0..1  (16-row tile)
  const int wn   = wave >> 1;       // 0..3  (16-col tile)
  const int i0   = blockIdx.x * 32; // permuted-sequence position base
  const int n0   = blockIdx.y * 64 + wn * 16;
  const int b    = blockIdx.z;

  // Stage 32 rows (gathered by perm) into LDS as bf16.
  for (int e = tid; e < 32 * HH; e += 256) {
    const int row = e >> 9, col = e & (HH - 1);
    const int tok = perm[i0 + row];
    xs[row * LDA + col] = f2bf(x[((size_t)b * SS + tok) * HH + col]);
  }
  __syncthreads();

  v8f accq = {}, acck = {}, accv = {};
  const unsigned short* abase = xs + (wm * 16) * LDA;
#pragma unroll 4
  for (int kk = 0; kk < HH / 32; ++kk) {
    const int k0 = kk * 32;
    const v16bf a = load_frag16(abase + k0, LDA, lane);
    const v16bf bqf = load_frag16(wq_bf + (size_t)n0 * HH + k0, HH, lane);
    const v16bf bkf = load_frag16(wk_bf + (size_t)n0 * HH + k0, HH, lane);
    const v16bf bvf = load_frag16(wv_bf + (size_t)n0 * HH + k0, HH, lane);
    if (kk + 1 < HH / 32)
      __builtin_prefetch(wq_bf + (size_t)n0 * HH + k0 + 32, 0, 3);
    accq = wmma_bf16(a, bqf, accq);
    acck = wmma_bf16(a, bkf, acck);
    accv = wmma_bf16(a, bvf, accv);
  }

  const int half = (lane >> 4) & 1;
  const int col  = n0 + (lane & 15);
  const float biasq = bq[col], biask = bk[col], biasv = bv[col];
#pragma unroll
  for (int j = 0; j < 8; ++j) {
    const int row = i0 + wm * 16 + half * 8 + j;   // permuted position
    const size_t off = ((size_t)b * SS + row) * HH + col;
    qp[off] = f2bf(accq[j] + biasq);
    kp[off] = f2bf(acck[j] + biask);
    vp[off] = f2bf(accv[j] + biasv);
  }
}

// ---------------------------------------------------------------------------
// Kernel 2: attention per (head, batch). 64 queries x 64 keys, feature 512.
//   Phase A: scores = Q K^T / sqrt(H)       (WMMA bf16, fp32 scores in LDS)
//   Phase B: row softmax (v_exp_f32), pack weights to bf16 in LDS
//   Phase C: attended = P V, scattered back through perm (inverse permute)
// ---------------------------------------------------------------------------
constexpr int LDSC = 68;  // fp32 score row stride
constexpr int LDSW = 72;  // bf16 weight row stride (144 B, 16B aligned)

__global__ __launch_bounds__(256) void k_attention(
    const unsigned short* __restrict__ qp, const unsigned short* __restrict__ kp,
    const unsigned short* __restrict__ vp, const int* __restrict__ perm,
    unsigned short* __restrict__ att) {
  __shared__ float          sc[TT * LDSC];
  __shared__ unsigned short pw[TT * LDSW];

  const int tid  = threadIdx.x;
  const int wave = tid >> 5;
  const int lane = tid & 31;
  const int half = (lane >> 4) & 1;
  const int c15  = lane & 15;
  const int head = blockIdx.x;
  const int b    = blockIdx.y;
  const size_t rowbase = ((size_t)b * SS + head * TT);

  // ---- Phase A: scores (4x4 tiles of 16x16; wave w owns tiles w and w+8,
  //      which share tn -> B fragment reused across both accumulators).
  {
    const int tn  = wave & 3;
    const int tm0 = wave >> 2;        // 0..1
    const int tm1 = tm0 + 2;          // 2..3
    v8f acc0 = {}, acc1 = {};
    const unsigned short* kbase = kp + (rowbase + tn * 16) * HH;
    const unsigned short* qbase0 = qp + (rowbase + tm0 * 16) * HH;
    const unsigned short* qbase1 = qp + (rowbase + tm1 * 16) * HH;
#pragma unroll 4
    for (int kk = 0; kk < HH / 32; ++kk) {
      const int k0 = kk * 32;
      const v16bf bf = load_frag16(kbase + k0, HH, lane);
      const v16bf a0 = load_frag16(qbase0 + k0, HH, lane);
      const v16bf a1 = load_frag16(qbase1 + k0, HH, lane);
      acc0 = wmma_bf16(a0, bf, acc0);
      acc1 = wmma_bf16(a1, bf, acc1);
    }
    const float scale = 0.044194173824159216f;   // 1/sqrt(512)
#pragma unroll
    for (int j = 0; j < 8; ++j) {
      sc[(tm0 * 16 + half * 8 + j) * LDSC + tn * 16 + c15] = acc0[j] * scale;
      sc[(tm1 * 16 + half * 8 + j) * LDSC + tn * 16 + c15] = acc1[j] * scale;
    }
  }
  __syncthreads();

  // ---- Phase B: softmax over each of the 64 rows (threads 0..63).
  if (tid < TT) {
    float* row = sc + tid * LDSC;
    float m = -3.4e38f;
#pragma unroll 8
    for (int c = 0; c < TT; ++c) m = fmaxf(m, row[c]);
    float s = 0.f;
#pragma unroll 8
    for (int c = 0; c < TT; ++c) { float e = __expf(row[c] - m); row[c] = e; s += e; }
    const float inv = __frcp_rn(s);
    unsigned short* wrow = pw + tid * LDSW;
#pragma unroll 8
    for (int c = 0; c < TT; ++c) wrow[c] = f2bf(row[c] * inv);
  }
  __syncthreads();

  // ---- Phase C: attended = P (64x64) @ V (64x512). Wave w: tm = w>>1,
  //      cols (w&1)*256 .. +255 in 16 tiles. A fragments hoisted (K=64 -> 2).
  {
    const int tm = wave >> 1;
    const v16bf aP0 = load_frag16(pw + tm * 16 * LDSW + 0,  LDSW, lane);
    const v16bf aP1 = load_frag16(pw + tm * 16 * LDSW + 32, LDSW, lane);

    int ptok[8];
#pragma unroll
    for (int j = 0; j < 8; ++j)
      ptok[j] = perm[head * TT + tm * 16 + half * 8 + j];

    for (int tt = 0; tt < 16; ++tt) {
      const int n0 = ((wave & 1) * 16 + tt) * 16;
      v8f acc = {};
      const v16bf b0 = load_frag16_strided(vp + rowbase * HH + n0, HH, lane);
      const v16bf b1 = load_frag16_strided(vp + (rowbase + 32) * HH + n0, HH, lane);
      acc = wmma_bf16(aP0, b0, acc);
      acc = wmma_bf16(aP1, b1, acc);
      const int col = n0 + c15;
#pragma unroll
      for (int j = 0; j < 8; ++j)   // inverse-permute scatter
        att[((size_t)b * SS + ptok[j]) * HH + col] = f2bf(acc[j]);
    }
  }
}

// ---------------------------------------------------------------------------
// Kernel 3: output projection. out = attended @ Wo^T + bo (fp32 out).
// ---------------------------------------------------------------------------
__global__ __launch_bounds__(256) void k_out_project(
    const unsigned short* __restrict__ att,
    const unsigned short* __restrict__ wo_bf, const float* __restrict__ bo,
    float* __restrict__ out) {
  __shared__ unsigned short as[32 * LDA];

  const int tid  = threadIdx.x;
  const int wave = tid >> 5;
  const int lane = tid & 31;
  const int wm   = wave & 1;
  const int wn   = wave >> 1;
  const int s0   = blockIdx.x * 32;
  const int n0   = blockIdx.y * 64 + wn * 16;
  const int b    = blockIdx.z;

  for (int e = tid; e < 32 * HH; e += 256) {
    const int row = e >> 9, col = e & (HH - 1);
    as[row * LDA + col] = att[((size_t)b * SS + s0 + row) * HH + col];
  }
  __syncthreads();

  v8f acc = {};
  const unsigned short* abase = as + (wm * 16) * LDA;
#pragma unroll 4
  for (int kk = 0; kk < HH / 32; ++kk) {
    const int k0 = kk * 32;
    const v16bf a  = load_frag16(abase + k0, LDA, lane);
    const v16bf bf = load_frag16(wo_bf + (size_t)n0 * HH + k0, HH, lane);
    if (kk + 1 < HH / 32)
      __builtin_prefetch(wo_bf + (size_t)n0 * HH + k0 + 32, 0, 3);
    acc = wmma_bf16(a, bf, acc);
  }

  const int half = (lane >> 4) & 1;
  const int col  = n0 + (lane & 15);
  const float bias = bo[col];
#pragma unroll
  for (int j = 0; j < 8; ++j) {
    const int row = s0 + wm * 16 + half * 8 + j;
    out[((size_t)b * SS + row) * HH + col] = acc[j] + bias;
  }
}

} // anonymous namespace

// ---------------------------------------------------------------------------
// Launch. Inputs: x, Wq, bq, Wk, bk, Wv, bv, Wo, bo, perm.
// Workspace: [Wq|Wk|Wv|Wo bf16 : 2 MB][Qp|Kp|Vp|att bf16 : 4 x 32 MB].
// ---------------------------------------------------------------------------
extern "C" void kernel_launch(void* const* d_in, const int* in_sizes, int n_in,
                              void* d_out, int out_size, void* d_ws, size_t ws_size,
                              hipStream_t stream) {
  (void)in_sizes; (void)n_in; (void)out_size; (void)ws_size;

  const float* x  = (const float*)d_in[0];
  const float* Wq = (const float*)d_in[1];
  const float* bq = (const float*)d_in[2];
  const float* Wk = (const float*)d_in[3];
  const float* bk = (const float*)d_in[4];
  const float* Wv = (const float*)d_in[5];
  const float* bv = (const float*)d_in[6];
  const float* Wo = (const float*)d_in[7];
  const float* bo = (const float*)d_in[8];
  const int*   perm = (const int*)d_in[9];
  float* out = (float*)d_out;

  unsigned short* wsbf = (unsigned short*)d_ws;
  unsigned short* wq_bf = wsbf;
  unsigned short* wk_bf = wsbf + WELT;
  unsigned short* wv_bf = wsbf + 2 * WELT;
  unsigned short* wo_bf = wsbf + 3 * WELT;
  unsigned short* qp  = wsbf + 4 * WELT;
  unsigned short* kp  = qp + QELT;
  unsigned short* vp  = kp + QELT;
  unsigned short* att = vp + QELT;

  k_convert_weights<<<dim3(256, 4), 256, 0, stream>>>(Wq, Wk, Wv, Wo, wsbf);

  k_qkv_project<<<dim3(SS / 32, HH / 64, BB), 256, 0, stream>>>(
      x, wq_bf, wk_bf, wv_bf, bq, bk, bv, perm, qp, kp, vp);

  k_attention<<<dim3(NH, BB), 256, 0, stream>>>(qp, kp, vp, perm, att);

  k_out_project<<<dim3(SS / 32, HH / 64, BB), 256, 0, stream>>>(
      att, wo_bf, bo, out);
}
